// SpatialGCN_42734924595842
// MI455X (gfx1250) — matrix operand order repaired and checked by reading, not compiled, not run on hardware
//
#include <hip/hip_runtime.h>

#define B_ 64
#define L_ 243
#define J_ 17
#define D_ 256
#define NFRAMES (B_*L_)            /* 15552 */
#define NROWS (NFRAMES*J_)         /* 264384 */
#define EPS_ 1e-5f

/* kernel1 tiling */
#define K1_FPB 2
#define K1_VROWS (K1_FPB*J_)       /* 34 valid rows */
#define K1_MT 3                    /* 3 row tiles -> 48 padded rows */
#define K1_ROWS 48
#define AS_S 518                   /* LDS stride (bf16 elems), conflict-free */
#define K1_BLOCKS (NFRAMES/K1_FPB) /* 7776 */

/* kernel3 tiling */
#define K3_ROWS 32
#define K3_BLOCKS (NROWS/K3_ROWS)  /* 8262 */
#define HN_S 262
#define MID_S 518

typedef __attribute__((ext_vector_type(16))) __bf16 bf16x16;
typedef __attribute__((ext_vector_type(8)))  float  f32x8;
typedef __attribute__((ext_vector_type(8)))  unsigned int u32x8;

__device__ __forceinline__ unsigned short f2bf(float f) {
  unsigned int u = __float_as_uint(f);
  u += 0x7fffu + ((u >> 16) & 1u);          /* round-to-nearest-even */
  return (unsigned short)(u >> 16);
}

__device__ __forceinline__ f32x8 wmma_bf16(u32x8 a, u32x8 b, f32x8 c) {
  return __builtin_amdgcn_wmma_f32_16x16x32_bf16(
      false, __builtin_bit_cast(bf16x16, a),
      false, __builtin_bit_cast(bf16x16, b),
      (short)0, c, false, false);
}

/* A fragment from LDS bf16 slab (row-major, stride in elements).
   16-bit A layout: lane m=lane&15 holds row, k = kt*32 + (v>>2)*16 + hi*8 + (v&3)*2 */
__device__ __forceinline__ u32x8 lds_a_frag(const unsigned short* base, int row0,
                                            int stride, int kt) {
  const int lane = threadIdx.x & 31;
  const int m = lane & 15, hi = lane >> 4;
  const unsigned short* rp = base + (row0 + m) * stride;
  u32x8 r;
#pragma unroll
  for (int v = 0; v < 8; ++v) {
    const int k = kt * 32 + ((v >> 2) << 4) + (hi << 3) + ((v & 3) << 1);
    r[v] = *reinterpret_cast<const unsigned int*>(rp + k);
  }
  return r;
}

/* B fragment: pre-swizzled in workspace so each lane reads 32 contiguous bytes */
__device__ __forceinline__ u32x8 ld_b_frag(const unsigned int* wf, int kt, int nt,
                                           int ntc) {
  const int lane = threadIdx.x & 31;
  const unsigned int* p = wf + ((((size_t)kt * ntc + nt) * 32 + lane) << 3);
  return *reinterpret_cast<const u32x8*>(p);
}

/* ---- kernel 0: repack f32 weights into bf16 WMMA B-fragment layout ----
   B layout (16-bit, 32x16): lane n=lane&15 is column, k = kt*32 + (v>>2)*16 + hi*8 + (v&3)*2 */
__global__ __launch_bounds__(256) void k0_repack(const float* __restrict__ src0,
                                                 const float* __restrict__ src1,
                                                 unsigned int* __restrict__ dst,
                                                 int ntc, int ktc, int rowlen, int ksw) {
  const int total = ntc * ktc * 256;
  for (int idx = blockIdx.x * 256 + threadIdx.x; idx < total; idx += gridDim.x * 256) {
    const int v = idx & 7;
    const int lane = (idx >> 3) & 31;
    const int t2 = idx >> 8;
    const int nt = t2 % ntc;
    const int kt = t2 / ntc;
    const int n = nt * 16 + (lane & 15);
    const int k = kt * 32 + ((v >> 2) << 4) + ((lane >> 4) << 3) + ((v & 3) << 1);
    const float* s = (k < ksw) ? (src0 + (size_t)n * rowlen + k)
                               : (src1 + (size_t)n * rowlen + (k - ksw));
    const unsigned int lo = f2bf(s[0]);
    const unsigned int hiw = f2bf(s[1]);
    dst[idx] = lo | (hiw << 16);
  }
}

/* ---- kernel 1: fused adjacency-folded dual projection + bias + ReLU + BN partials
   h = (adjT X)W1^T + (adj X)W2^T + rs1*b1 + rs2*b2 ; single K=512 WMMA chain ---- */
__global__ __launch_bounds__(256) void k1_gcn(const float* __restrict__ x,
                                              const float* __restrict__ adj,
                                              const float* __restrict__ b1,
                                              const float* __restrict__ b2,
                                              const unsigned int* __restrict__ wcatf,
                                              float* __restrict__ hbuf,
                                              float* __restrict__ part) {
  __shared__ unsigned short As[K1_ROWS * AS_S];
  __shared__ float adjS[J_ * J_];
  __shared__ float rs1[J_], rs2[J_];
  __shared__ float red[512];
  const int tid = threadIdx.x;
  const int g = blockIdx.x;

  for (int t = tid; t < J_ * J_; t += 256) adjS[t] = adj[t];
  __syncthreads();
  if (tid < J_) {
    float s1 = 0.f, s2 = 0.f;
    for (int j = 0; j < J_; ++j) { s1 += adjS[j * J_ + tid]; s2 += adjS[tid * J_ + j]; }
    rs1[tid] = s1; rs2[tid] = s2;
  }
  /* zero pad rows 34..47 (k 0..511) */
  for (int t = tid; t < (K1_ROWS - K1_VROWS) * 512; t += 256) {
    const int r = K1_VROWS + (t >> 9);
    As[r * AS_S + (t & 511)] = 0;
  }
  /* build A = [adjT X | adj X] in bf16, float4 at a time */
  for (int idx = tid; idx < K1_VROWS * 128; idx += 256) {
    const int c5 = (idx & 127) << 2;          /* 0..508 step 4 */
    const int fi = idx >> 7;                  /* 0..33 */
    const int f = (fi >= J_) ? 1 : 0;
    const int i = fi - f * J_;
    const int half = c5 >> 8;
    const int c = c5 & 255;
    const size_t base = ((size_t)g * K1_VROWS + f * J_) * D_ + c;
    float s0 = 0.f, s1 = 0.f, s2 = 0.f, s3 = 0.f;
#pragma unroll
    for (int j = 0; j < J_; ++j) {
      const float a = half ? adjS[i * J_ + j] : adjS[j * J_ + i];
      const float4 xv = *reinterpret_cast<const float4*>(x + base + (size_t)j * D_);
      s0 += a * xv.x; s1 += a * xv.y; s2 += a * xv.z; s3 += a * xv.w;
    }
    unsigned int* wp = reinterpret_cast<unsigned int*>(As + fi * AS_S + c5);
    wp[0] = (unsigned int)f2bf(s0) | ((unsigned int)f2bf(s1) << 16);
    wp[1] = (unsigned int)f2bf(s2) | ((unsigned int)f2bf(s3) << 16);
  }
  __syncthreads();

  const int wave = tid >> 5, lane = tid & 31, m = lane & 15, hi = lane >> 4;
  for (int jj = 0; jj < 2; ++jj) {
    const int nt = wave * 2 + jj;
    f32x8 acc[K1_MT];
#pragma unroll
    for (int mt = 0; mt < K1_MT; ++mt) acc[mt] = (f32x8){0,0,0,0,0,0,0,0};
    for (int kt = 0; kt < 16; ++kt) {
      const u32x8 bf = ld_b_frag(wcatf, kt, nt, 16);
#pragma unroll
      for (int mt = 0; mt < K1_MT; ++mt) {
        const u32x8 af = lds_a_frag(As, mt * 16, AS_S, kt);
        acc[mt] = wmma_bf16(af, bf, acc[mt]);
      }
    }
    const int ch = nt * 16 + m;
    const float b1n = b1[ch], b2n = b2[ch];
    float bs = 0.f, bq = 0.f;
#pragma unroll
    for (int mt = 0; mt < K1_MT; ++mt) {
#pragma unroll
      for (int v = 0; v < 8; ++v) {
        const int fi = mt * 16 + v + hi * 8;
        if (fi < K1_VROWS) {
          const int i = (fi >= J_) ? fi - J_ : fi;
          float h = acc[mt][v] + rs1[i] * b1n + rs2[i] * b2n;
          h = fmaxf(h, 0.f);
          hbuf[((size_t)g * K1_VROWS + fi) * D_ + ch] = h;
          bs += h; bq += h * h;
        }
      }
    }
    red[tid] = bs; red[256 + tid] = bq;
    __syncthreads();
    if (lane < 16) {
      part[(size_t)g * 512 + ch]       = red[tid] + red[tid + 16];
      part[(size_t)g * 512 + 256 + ch] = red[256 + tid] + red[256 + tid + 16];
    }
    __syncthreads();
  }
}

/* ---- kernel 2a/2b: deterministic two-level BN stat reduction ---- */
__global__ __launch_bounds__(256) void k2a(const float* __restrict__ part,
                                           float* __restrict__ part2) {
  const int p = blockIdx.x, t = threadIdx.x;
  const int chunk = K1_BLOCKS / 32;  /* 243 */
  float s0 = 0.f, s1 = 0.f;
  for (int b = p * chunk; b < (p + 1) * chunk; ++b) {
    s0 += part[(size_t)b * 512 + t];
    s1 += part[(size_t)b * 512 + 256 + t];
  }
  part2[(size_t)p * 512 + t] = s0;
  part2[(size_t)p * 512 + 256 + t] = s1;
}

__global__ __launch_bounds__(256) void k2b(const float* __restrict__ part2,
                                           const float* __restrict__ gam,
                                           const float* __restrict__ bet,
                                           float* __restrict__ bnA,
                                           float* __restrict__ bnB) {
  const int c = threadIdx.x;
  float s = 0.f, q = 0.f;
  for (int p = 0; p < 32; ++p) {
    s += part2[(size_t)p * 512 + c];
    q += part2[(size_t)p * 512 + 256 + c];
  }
  const float invN = 1.f / (float)NROWS;
  const float mean = s * invN;
  const float var = fmaxf(q * invN - mean * mean, 0.f);
  const float a = gam[c] * rsqrtf(var + EPS_);
  bnA[c] = a;
  bnB[c] = bet[c] - mean * a;
}

/* ---- kernel 3: BN + LN + MLP (two WMMA GEMMs), in-place over h rows ---- */
__global__ __launch_bounds__(128) void k3_mlp(const float* __restrict__ bnA,
                                              const float* __restrict__ bnB,
                                              const float* __restrict__ lng,
                                              const float* __restrict__ lnb,
                                              const unsigned int* __restrict__ mw1f,
                                              const float* __restrict__ mb1,
                                              const unsigned int* __restrict__ mw2f,
                                              const float* __restrict__ mb2,
                                              float* __restrict__ hbuf) {
  __shared__ unsigned short hn[K3_ROWS * HN_S];
  __shared__ unsigned short mid[K3_ROWS * MID_S];
  __shared__ float lnred[256];
  float* vtmp = reinterpret_cast<float*>(mid);  /* 32*256 f32 fits in mid */
  const int tid = threadIdx.x;
  const size_t R0 = (size_t)blockIdx.x * K3_ROWS;
  const int row = tid >> 2, q = tid & 3, c0 = q * 64;

  /* stage: BN apply + LN partials */
  float s = 0.f, sq = 0.f;
  const float* hrow = hbuf + (R0 + row) * (size_t)D_;
#pragma unroll
  for (int cc = 0; cc < 16; ++cc) {
    const int c = c0 + cc * 4;
    const float4 h4 = *reinterpret_cast<const float4*>(hrow + c);
    const float v0 = bnA[c + 0] * h4.x + bnB[c + 0];
    const float v1 = bnA[c + 1] * h4.y + bnB[c + 1];
    const float v2 = bnA[c + 2] * h4.z + bnB[c + 2];
    const float v3 = bnA[c + 3] * h4.w + bnB[c + 3];
    vtmp[row * D_ + c + 0] = v0; vtmp[row * D_ + c + 1] = v1;
    vtmp[row * D_ + c + 2] = v2; vtmp[row * D_ + c + 3] = v3;
    s += v0 + v1 + v2 + v3;
    sq += v0 * v0 + v1 * v1 + v2 * v2 + v3 * v3;
  }
  lnred[tid] = s;
  __syncthreads();
  const int b0 = tid & ~3;
  const float ls = lnred[b0] + lnred[b0 + 1] + lnred[b0 + 2] + lnred[b0 + 3];
  __syncthreads();
  lnred[tid] = sq;
  __syncthreads();
  const float lq = lnred[b0] + lnred[b0 + 1] + lnred[b0 + 2] + lnred[b0 + 3];
  const float lmu = ls * (1.f / 256.f);
  const float lvar = fmaxf(lq * (1.f / 256.f) - lmu * lmu, 0.f);
  const float lrs = rsqrtf(lvar + EPS_);
#pragma unroll
  for (int cc = 0; cc < 64; ++cc) {
    const int c = c0 + cc;
    const float v = vtmp[row * D_ + c];
    hn[row * HN_S + c] = f2bf((v - lmu) * lrs * lng[c] + lnb[c]);
  }
  __syncthreads();

  const int wave = tid >> 5, lane = tid & 31, m = lane & 15, hi = lane >> 4;
  /* GEMM1: [32x256] x [512x256]^T -> ReLU -> mid bf16 */
  for (int job = wave; job < 64; job += 4) {
    const int mt = job & 1, nt = job >> 1;
    f32x8 acc = (f32x8){0,0,0,0,0,0,0,0};
    for (int kt = 0; kt < 8; ++kt) {
      const u32x8 bf = ld_b_frag(mw1f, kt, nt, 32);
      const u32x8 af = lds_a_frag(hn, mt * 16, HN_S, kt);
      acc = wmma_bf16(af, bf, acc);
    }
    const float bias = mb1[nt * 16 + m];
#pragma unroll
    for (int v = 0; v < 8; ++v) {
      const float r = fmaxf(acc[v] + bias, 0.f);
      mid[(mt * 16 + v + hi * 8) * MID_S + nt * 16 + m] = f2bf(r);
    }
  }
  __syncthreads();
  /* GEMM2: [32x512] x [256x512]^T -> out (in place over consumed h rows) */
  for (int job = wave; job < 32; job += 4) {
    const int mt = job & 1, nt = job >> 1;
    f32x8 acc = (f32x8){0,0,0,0,0,0,0,0};
    for (int kt = 0; kt < 16; ++kt) {
      const u32x8 bf = ld_b_frag(mw2f, kt, nt, 16);
      const u32x8 af = lds_a_frag(mid, mt * 16, MID_S, kt);
      acc = wmma_bf16(af, bf, acc);
    }
    const float bias = mb2[nt * 16 + m];
#pragma unroll
    for (int v = 0; v < 8; ++v) {
      hbuf[(R0 + mt * 16 + v + hi * 8) * (size_t)D_ + nt * 16 + m] = acc[v] + bias;
    }
  }
}

extern "C" void kernel_launch(void* const* d_in, const int* in_sizes, int n_in,
                              void* d_out, int out_size, void* d_ws, size_t ws_size,
                              hipStream_t stream) {
  (void)in_sizes; (void)n_in; (void)out_size; (void)ws_size;
  const float* x   = (const float*)d_in[0];
  const float* adj = (const float*)d_in[1];
  const float* w1  = (const float*)d_in[2];
  const float* b1  = (const float*)d_in[3];
  const float* w2  = (const float*)d_in[4];
  const float* b2  = (const float*)d_in[5];
  const float* bng = (const float*)d_in[6];
  const float* bnb = (const float*)d_in[7];
  const float* lng = (const float*)d_in[8];
  const float* lnb = (const float*)d_in[9];
  const float* mw1 = (const float*)d_in[10];
  const float* mb1 = (const float*)d_in[11];
  const float* mw2 = (const float*)d_in[12];
  const float* mb2 = (const float*)d_in[13];

  char* ws = (char*)d_ws;
  unsigned int* wcatf = (unsigned int*)(ws + 0);        /* 256 KB: [W1|W2] K=512 */
  unsigned int* mw1f  = (unsigned int*)(ws + 262144);   /* 256 KB */
  unsigned int* mw2f  = (unsigned int*)(ws + 524288);   /* 256 KB */
  float* bnA  = (float*)(ws + 786432);
  float* bnB  = (float*)(ws + 787456);
  float* part = (float*)(ws + 788480);                  /* 7776*512 f32 */
  float* part2 = (float*)(ws + 788480 + (size_t)K1_BLOCKS * 512 * 4);
  float* hbuf = (float*)d_out;                          /* reused as h scratch */

  k0_repack<<<256, 256, 0, stream>>>(w1, w2, wcatf, 16, 16, 256, 256);
  k0_repack<<<256, 256, 0, stream>>>(mw1, mw1, mw1f, 32, 8, 256, 256);
  k0_repack<<<256, 256, 0, stream>>>(mw2, mw2, mw2f, 16, 16, 512, 512);
  k1_gcn<<<K1_BLOCKS, 256, 0, stream>>>(x, adj, b1, b2, wcatf, hbuf, part);
  k2a<<<32, 256, 0, stream>>>(part, part2);
  k2b<<<1, 256, 0, stream>>>(part2, bng, bnb, bnA, bnB);
  k3_mlp<<<K3_BLOCKS, 128, 0, stream>>>(bnA, bnB, lng, lnb, mw1f, mb1, mw2f, mb2, hbuf);
}
